// hd_id_lvl_decoder_73581379715548
// MI455X (gfx1250) — matrix operand order repaired and checked by reading, not compiled, not run on hardware
//
#include <hip/hip_runtime.h>

#define D_TOT   10000
#define NUM_ID  256
#define NUM_LVL 64
#define DK      128                    // D-chunk staged in LDS per iteration
#define LSTRIDE (DK + 8)               // bf16 elems per LDS row; +16B pad -> conflict-free
#define BIN_LEN 0.015625f

typedef __attribute__((ext_vector_type(16))) __bf16        v16bf;
typedef __attribute__((ext_vector_type(8)))  float         v8f;
typedef __attribute__((ext_vector_type(4)))  float         v4f;
typedef __attribute__((ext_vector_type(4)))  unsigned int  v4u;
typedef __attribute__((ext_vector_type(2)))  unsigned int  v2u;

// ---------------------------------------------------------------- helpers ----

// round-to-nearest-even fp32 -> bf16, packed pair (elem0 in low half)
__device__ __forceinline__ unsigned int pack2_bf16(float a, float b) {
    unsigned int ua = __builtin_bit_cast(unsigned int, a);
    unsigned int ub = __builtin_bit_cast(unsigned int, b);
    ua += 0x7FFFu + ((ua >> 16) & 1u);
    ub += 0x7FFFu + ((ub >> 16) & 1u);
    return (ua >> 16) | (ub & 0xFFFF0000u);
}

// unpack a bf16 pair to two fp32
__device__ __forceinline__ float bf16_lo(unsigned int u) {
    return __builtin_bit_cast(float, u << 16);
}
__device__ __forceinline__ float bf16_hi(unsigned int u) {
    return __builtin_bit_cast(float, u & 0xFFFF0000u);
}

// A-matrix (16x32 bf16) fragment: lane L holds row (L&15);
// elems 0..7  -> K = 8*(L>>4) + 0..7
// elems 8..15 -> K = 8*(L>>4) + 16..23           (ISA 05_wmma 16-bit A layout)
__device__ __forceinline__ v16bf load_frag_a(const unsigned short* lds, int row,
                                             int kw, int lane) {
    const int base = (lane >> 4) * 8;
    const unsigned short* p = lds + row * LSTRIDE + kw + base;
    union { v4u q[2]; v16bf v; } r;
    r.q[0] = *(const v4u*)(p);        // ds_load_b128
    r.q[1] = *(const v4u*)(p + 16);   // ds_load_b128
    return r.v;
}

// B-matrix (32x16 bf16) fragment: lane L holds column (L&15);
// elems 0..7 -> K = 8*(L>>4) + 0..7, elems 8..15 -> K = 8*(L>>4) + 16..23 (matches A)
__device__ __forceinline__ v16bf load_frag_b(const unsigned short* lds, int row,
                                             int kw, int lane) {
    const int base = (lane >> 4) * 8;
    const unsigned short* p = lds + row * LSTRIDE + kw + base;
    union { v4u q[2]; v16bf v; } r;
    r.q[0] = *(const v4u*)(p);
    r.q[1] = *(const v4u*)(p + 16);
    return r.v;
}

// WMMA mainloop + argmax epilogue shared by both kernels.
// Alds: 128 id-rows of bf16(x*id); Blds: 64 lvl-rows of bf16(lvl).
__device__ __forceinline__ void wmma_step(const unsigned short* Alds,
                                          const unsigned short* Blds,
                                          int wave, int lane, v8f acc[2][4]) {
    const int hl    = lane & 15;
    const int mrow0 = wave * 32 + hl;
    const int mrow1 = mrow0 + 16;
    #pragma unroll
    for (int ks = 0; ks < DK; ks += 32) {
        const v16bf a0 = load_frag_a(Alds, mrow0, ks, lane);
        const v16bf a1 = load_frag_a(Alds, mrow1, ks, lane);
        #pragma unroll
        for (int nt = 0; nt < 4; ++nt) {
            const v16bf bm = load_frag_b(Blds, nt * 16 + hl, ks, lane);
            acc[0][nt] = __builtin_amdgcn_wmma_f32_16x16x32_bf16(
                false, a0, false, bm, (short)0, acc[0][nt], false, false);
            acc[1][nt] = __builtin_amdgcn_wmma_f32_16x16x32_bf16(
                false, a1, false, bm, (short)0, acc[1][nt], false, false);
        }
    }
}

// argmax over 64 levels.  C/D layout: lane holds level n = (lane&15) + 16*nt,
// VGPR v holds row M = v + 8*(lane>>4).  First-max tie-break (== jnp.argmax).
__device__ __forceinline__ void argmax_epilogue(v8f acc[2][4], float* out,
                                                int b, int i0, int wave, int lane) {
    const int hl = lane & 15;
    #pragma unroll
    for (int mt = 0; mt < 2; ++mt) {
        #pragma unroll
        for (int v = 0; v < 8; ++v) {
            float bestv = acc[mt][0][v];
            int   bestl = hl;
            #pragma unroll
            for (int nt = 1; nt < 4; ++nt) {
                const float s = acc[mt][nt][v];
                const int   l = nt * 16 + hl;
                if (s > bestv) { bestv = s; bestl = l; }
            }
            #pragma unroll
            for (int mask = 8; mask >= 1; mask >>= 1) {   // stays in 16-lane half
                const float ov = __shfl_xor(bestv, mask, 32);
                const int   ol = __shfl_xor(bestl, mask, 32);
                if (ov > bestv || (ov == bestv && ol < bestl)) { bestv = ov; bestl = ol; }
            }
            if (hl == 0) {
                const int rowm = v + 8 * (lane >> 4);
                out[(size_t)b * NUM_ID + i0 + wave * 32 + mt * 16 + rowm] =
                    (float)bestl * BIN_LEN;
            }
        }
    }
}

// ------------------------------------------------- kernel 0: fp32 -> bf16 ----

__global__ __launch_bounds__(256)
void cvt_bf16_kernel(const float* __restrict__ src,
                     unsigned short* __restrict__ dst, int n4) {
    const int i = blockIdx.x * blockDim.x + threadIdx.x;
    if (i < n4) {
        const v4f v = *(const v4f*)(src + (size_t)i * 4);
        v2u p = { pack2_bf16(v.x, v.y), pack2_bf16(v.z, v.w) };
        *(v2u*)(dst + (size_t)i * 4) = p;
    }
}

// ------------------------------ main kernel, bf16-fed (needs 6.4 MB of ws) ----

__global__ __launch_bounds__(128)
void hd_decoder_bf16_kernel(const float* __restrict__ x,
                            const unsigned short* __restrict__ id_bf,   // packed bf16
                            const unsigned short* __restrict__ lvl_bf,  // packed bf16
                            float* __restrict__ out) {
    __shared__ unsigned short Alds[128 * LSTRIDE]; // bf16(x[b,d] * id[i,d]), 128 ids
    __shared__ unsigned short Blds[ 64 * LSTRIDE]; // bf16(lvl[l,d]),          64 lvls

    const int b    = blockIdx.x >> 1;
    const int i0   = (blockIdx.x & 1) * 128;
    const int t    = threadIdx.x;
    const int lane = t & 31;
    const int wave = t >> 5;

    // fill coords: thread owns 8 consecutive d-columns (one v4u of bf16 pairs)
    const int dcol   = (t & 15) * 8;           // 0..120
    const int rgroup = t >> 4;                 // 0..7
    const int raseA  = rgroup * 16;            // 16 A-rows per thread
    const int raseB  = rgroup * 8;             // 8  B-rows per thread

    v8f acc[2][4] = {};

    for (int chunk = 0; chunk < D_TOT; chunk += DK) {
        const int  d   = chunk + dcol;
        const bool inb = (d < D_TOT);          // D_TOT % 8 == 0 -> whole-v4u guard

        if (chunk + DK < D_TOT) {              // global_prefetch_b8 on next chunk
            __builtin_prefetch(x      + (size_t)b * D_TOT + chunk + DK + dcol, 0, 1);
            __builtin_prefetch(id_bf  + (size_t)(i0 + raseA) * D_TOT + chunk + DK + dcol, 0, 1);
            __builtin_prefetch(lvl_bf + (size_t)raseB * D_TOT + chunk + DK + dcol, 0, 1);
        }

        float xs[8] = {};
        if (inb) {
            const v4f xl = *(const v4f*)(x + (size_t)b * D_TOT + d);
            const v4f xh = *(const v4f*)(x + (size_t)b * D_TOT + d + 4);
            xs[0]=xl.x; xs[1]=xl.y; xs[2]=xl.z; xs[3]=xl.w;
            xs[4]=xh.x; xs[5]=xh.y; xs[6]=xh.z; xs[7]=xh.w;
        }

        // A fill: bf16 id * fp32 x -> bf16 product, one b128 load + b128 store/row
        #pragma unroll 4
        for (int r = 0; r < 16; ++r) {
            const int row = raseA + r;
            v4u idq = {0u, 0u, 0u, 0u};
            if (inb) idq = *(const v4u*)(id_bf + (size_t)(i0 + row) * D_TOT + d);
            v4u pq;
            #pragma unroll
            for (int j = 0; j < 4; ++j) {
                const float p0 = bf16_lo(idq[j]) * xs[2 * j];
                const float p1 = bf16_hi(idq[j]) * xs[2 * j + 1];
                pq[j] = pack2_bf16(p0, p1);
            }
            *(v4u*)(Alds + row * LSTRIDE + dcol) = pq;
        }
        // B fill: straight bf16 copy, zero VALU
        #pragma unroll
        for (int r = 0; r < 8; ++r) {
            const int row = raseB + r;
            v4u lq = {0u, 0u, 0u, 0u};
            if (inb) lq = *(const v4u*)(lvl_bf + (size_t)row * D_TOT + d);
            *(v4u*)(Blds + row * LSTRIDE + dcol) = lq;
        }

        __syncthreads();
        wmma_step(Alds, Blds, wave, lane, acc);
        __syncthreads();
    }

    argmax_epilogue(acc, out, b, i0, wave, lane);
}

// -------------------- fallback: fused fp32-fed kernel (no workspace needed) ----

__global__ __launch_bounds__(128)
void hd_decoder_fused_kernel(const float* __restrict__ x,
                             const float* __restrict__ id_hvs,
                             const float* __restrict__ lvl_hvs,
                             float* __restrict__ out) {
    __shared__ unsigned short Alds[128 * LSTRIDE];
    __shared__ unsigned short Blds[ 64 * LSTRIDE];

    const int b    = blockIdx.x >> 1;
    const int i0   = (blockIdx.x & 1) * 128;
    const int t    = threadIdx.x;
    const int lane = t & 31;
    const int wave = t >> 5;

    const int dcol   = (t & 31) * 4;
    const int rgroup = t >> 5;
    const int raseA  = rgroup * 32;
    const int raseB  = rgroup * 16;

    v8f acc[2][4] = {};

    for (int chunk = 0; chunk < D_TOT; chunk += DK) {
        const int  d   = chunk + dcol;
        const bool inb = (d < D_TOT);

        v4f x4 = {0.f, 0.f, 0.f, 0.f};
        if (inb) x4 = *(const v4f*)(x + (size_t)b * D_TOT + d);

        #pragma unroll 4
        for (int r = 0; r < 32; ++r) {
            const int row = raseA + r;
            v4f id4 = {0.f, 0.f, 0.f, 0.f};
            if (inb) id4 = *(const v4f*)(id_hvs + (size_t)(i0 + row) * D_TOT + d);
            const v4f p = x4 * id4;
            v2u packed = { pack2_bf16(p.x, p.y), pack2_bf16(p.z, p.w) };
            *(v2u*)(Alds + row * LSTRIDE + dcol) = packed;
        }
        #pragma unroll 4
        for (int r = 0; r < 16; ++r) {
            const int row = raseB + r;
            v4f l4 = {0.f, 0.f, 0.f, 0.f};
            if (inb) l4 = *(const v4f*)(lvl_hvs + (size_t)row * D_TOT + d);
            v2u packed = { pack2_bf16(l4.x, l4.y), pack2_bf16(l4.z, l4.w) };
            *(v2u*)(Blds + row * LSTRIDE + dcol) = packed;
        }

        __syncthreads();
        wmma_step(Alds, Blds, wave, lane, acc);
        __syncthreads();
    }

    argmax_epilogue(acc, out, b, i0, wave, lane);
}

// ------------------------------------------------------------------ launch ----

extern "C" void kernel_launch(void* const* d_in, const int* in_sizes, int n_in,
                              void* d_out, int out_size, void* d_ws, size_t ws_size,
                              hipStream_t stream) {
    const float* x       = (const float*)d_in[0];
    const float* id_hvs  = (const float*)d_in[1];
    const float* lvl_hvs = (const float*)d_in[2];
    float* out = (float*)d_out;

    const int B = in_sizes[0] / D_TOT;          // 64
    dim3 grid(B * (NUM_ID / 128));              // 128 blocks
    dim3 block(128);

    const size_t need = (size_t)(NUM_ID + NUM_LVL) * D_TOT * sizeof(unsigned short);
    if (ws_size >= need && d_ws) {
        unsigned short* id_bf  = (unsigned short*)d_ws;
        unsigned short* lvl_bf = id_bf + (size_t)NUM_ID * D_TOT;

        const int n4_id  = NUM_ID  * D_TOT / 4;
        const int n4_lvl = NUM_LVL * D_TOT / 4;
        cvt_bf16_kernel<<<(n4_id  + 255) / 256, 256, 0, stream>>>(id_hvs,  id_bf,  n4_id);
        cvt_bf16_kernel<<<(n4_lvl + 255) / 256, 256, 0, stream>>>(lvl_hvs, lvl_bf, n4_lvl);

        hd_decoder_bf16_kernel<<<grid, block, 0, stream>>>(x, id_bf, lvl_bf, out);
    } else {
        hd_decoder_fused_kernel<<<grid, block, 0, stream>>>(x, id_hvs, lvl_hvs, out);
    }
}
